// StructureLayer_7756710937195
// MI455X (gfx1250) — matrix operand order repaired and checked by reading, not compiled, not run on hardware
//
#include <hip/hip_runtime.h>
#include <hip/hip_bf16.h>

// ---------------- problem constants ----------------
#define NSEQ 512
#define CS   384
#define CP   128
#define CH   16
#define NH   12
#define PQK  4
#define PV   8
#define DPROJ 1152     // 192(q) + 384(kv) + 144(qp) + 432(kvp)
#define DCAT  2112     // 192(o) + 288(op) + 96(|op|) + 1536(o_pair)

typedef __attribute__((ext_vector_type(16))) __bf16 v16bf;
typedef __attribute__((ext_vector_type(8)))  float  v8f;

// ---------------- WMMA fragment loaders (wave32, 16x16x32 bf16) ----------------
// A (16xK, row-major, lda=K): lane L -> row = L&15,
//   elems 0..15 : K = kb + (L>=16 ? 16:0) + e
__device__ inline v16bf load_a_frag(const __bf16* A, int lda, int row0, int kb, int lane) {
  const __bf16* p = A + (size_t)(row0 + (lane & 15)) * lda + kb + ((lane >> 4) * 16);
  v16bf r;
#pragma unroll
  for (int e = 0; e < 16; ++e) r[e] = p[e];
  return r;
}
// Same A layout but sourced from an fp32 buffer (LDS slab), converting on load.
__device__ inline v16bf load_a_frag_f32(const float* A, int lda, int row0, int kb, int lane) {
  const float* p = A + (size_t)(row0 + (lane & 15)) * lda + kb + ((lane >> 4) * 16);
  v16bf r;
#pragma unroll
  for (int e = 0; e < 16; ++e) r[e] = (__bf16)p[e];
  return r;
}
// B stored as B^T (N x K, row-major, ldb=K): lane L -> col = col0 + (L&15),
// holds 16 consecutive K starting at kb + (L>=16 ? 16 : 0).
__device__ inline v16bf load_b_frag(const __bf16* BT, int ldb, int col0, int kb, int lane) {
  const __bf16* p = BT + (size_t)(col0 + (lane & 15)) * ldb + kb + ((lane >> 4) * 16);
  v16bf r;
#pragma unroll
  for (int e = 0; e < 16; ++e) r[e] = p[e];
  return r;
}

// ---------------- async global -> LDS staging (CDNA5, ASYNCcnt tracked) ----------------
__device__ inline void async_load_b128_to_lds(unsigned lds_off, const void* gaddr) {
  unsigned long long ga = (unsigned long long)(uintptr_t)gaddr;
  asm volatile("global_load_async_to_lds_b128 %0, %1, off"
               :: "v"(lds_off), "v"(ga) : "memory");
}
__device__ inline void wait_async_zero() {
  asm volatile("s_wait_asynccnt 0x0" ::: "memory");
}

// ---------------- prep kernels ----------------
__global__ void convert_to_bf16_k(const float* __restrict__ src, __bf16* __restrict__ dst, int count) {
  int i = blockIdx.x * 256 + threadIdx.x;
  if (i < count) dst[i] = (__bf16)src[i];
}

// src is (K x Nc) row-major fp32; dst is (NcPad x K) bf16 (B^T layout), zero-padded cols.
__global__ void transpose_to_bf16_k(const float* __restrict__ src, __bf16* __restrict__ dst,
                                    int K, int Nc, int NcPad) {
  int idx = blockIdx.x * 256 + threadIdx.x;
  if (idx >= NcPad * K) return;
  int n = idx / K, k = idx % K;
  float v = (n < Nc) ? src[(size_t)k * Nc + n] : 0.f;
  dst[idx] = (__bf16)v;
}

__global__ void concat_bias_k(const float* b0, const float* b1, const float* b2, const float* b3,
                              float* dst) {
  int t = blockIdx.x * 256 + threadIdx.x;
  if (t >= DPROJ) return;
  float v;
  if (t < 192)      v = b0[t];
  else if (t < 576) v = b1[t - 192];
  else if (t < 720) v = b2[t - 576];
  else              v = b3[t - 720];
  dst[t] = v;
}

// ---------------- generic bf16 WMMA GEMM:  C = A@B + bias (+resid) (+relu) ----------------
// A: (M x K) bf16 row-major.  BT: (Nc x K) bf16 (i.e., B transposed).  C fp32, Cbf optional.
__global__ __launch_bounds__(256) void gemm_bf16_k(
    const __bf16* __restrict__ A, const __bf16* __restrict__ BT,
    const float* __restrict__ bias, const float* __restrict__ resid,
    float* __restrict__ C, __bf16* __restrict__ Cbf,
    int M, int Nc, int K, int relu) {
  int lane = threadIdx.x & 31, wave = threadIdx.x >> 5;
  int m0 = blockIdx.x * 16;
  int n0 = (blockIdx.y * 8 + wave) * 16;
  if (m0 >= M || n0 >= Nc) return;
  v8f acc = {};
  for (int kb = 0; kb < K; kb += 32) {
    v16bf a = load_a_frag(A, K, m0, kb, lane);
    v16bf b = load_b_frag(BT, K, n0, kb, lane);
    acc = __builtin_amdgcn_wmma_f32_16x16x32_bf16(false, a, false, b, (short)0, acc, false, false);
  }
  int col  = n0 + (lane & 15);
  int roff = (lane >> 4) * 8;
  float bv = bias ? bias[col] : 0.f;
#pragma unroll
  for (int r = 0; r < 8; ++r) {
    int row = m0 + roff + r;
    float v = acc[r] + bv;
    if (resid) v += resid[(size_t)row * Nc + col];
    if (relu)  v = fmaxf(v, 0.f);
    C[(size_t)row * Nc + col] = v;
    if (Cbf) Cbf[(size_t)row * Nc + col] = (__bf16)v;
  }
}

// ---------------- rotate/translate the point projections ----------------
__global__ void rotate_points_k(const float* __restrict__ proj, const float* __restrict__ rot,
                                const float* __restrict__ trans,
                                float* __restrict__ qp_r, float* __restrict__ kp_r,
                                float* __restrict__ vp_r) {
  int n = blockIdx.x, t = threadIdx.x;
  __shared__ float R[9], T[3];
  if (t < 9) R[t] = rot[n * 9 + t];
  if (t < 3) T[t] = trans[n * 3 + t];
  __syncthreads();
  if (t < NH * PQK) {  // query points: cols 576 + t*3
    const float* s = proj + (size_t)n * DPROJ + 576 + t * 3;
    float x = s[0], y = s[1], z = s[2];
    float* d = qp_r + ((size_t)n * NH * PQK + t) * 3;
    d[0] = R[0]*x + R[1]*y + R[2]*z + T[0];
    d[1] = R[3]*x + R[4]*y + R[5]*z + T[1];
    d[2] = R[6]*x + R[7]*y + R[8]*z + T[2];
  } else if (t < NH * PQK + NH * (PQK + PV)) {  // k/v points: cols 720 + idx*3
    int idx = t - NH * PQK;  // pidx = h*12 + pw
    const float* s = proj + (size_t)n * DPROJ + 720 + idx * 3;
    float x = s[0], y = s[1], z = s[2];
    float ox = R[0]*x + R[1]*y + R[2]*z + T[0];
    float oy = R[3]*x + R[4]*y + R[5]*z + T[1];
    float oz = R[6]*x + R[7]*y + R[8]*z + T[2];
    int h = idx / 12, pw = idx % 12;
    float* d = (pw < PQK) ? kp_r + (((size_t)n * NH + h) * PQK + pw) * 3
                          : vp_r + (((size_t)n * NH + h) * PV + (pw - PQK)) * 3;
    d[0] = ox; d[1] = oy; d[2] = oz;
  }
}

// ---------------- the fused IPA attention kernel (1 block per query i) ----------------
// dyn-LDS: p_lds[512*128]f32 | a_f[12*512]f32 | a_bf[16*512]bf16 | qv[192] qpv[144] hw[12] misc[16]
#define P_LDS_BYTES (NSEQ * CP * 4)                      // 262144
#define ATTN_SMEM (P_LDS_BYTES + 24576 + 16384 + (192 + 144 + 12 + 16) * 4)   // ~298 KB < 320 KB WGP LDS

__global__ __launch_bounds__(256) void ipa_attn_k(
    const float* __restrict__ proj, const float* __restrict__ p,
    const __bf16* __restrict__ wbT,
    const float* __restrict__ rot, const float* __restrict__ trans,
    const float* __restrict__ mask, const float* __restrict__ head_w,
    const float* __restrict__ qp_r, const float* __restrict__ kp_r,
    const float* __restrict__ vp_r,
    __bf16* __restrict__ cat_bf) {
  extern __shared__ char smem[];
  float*  p_lds = (float*)smem;                               // 512 x 128 fp32 (p[i] slab)
  float*  a_f   = (float*)(smem + P_LDS_BYTES);               // 12 x 512 logits/probs
  __bf16* a_bf  = (__bf16*)(smem + P_LDS_BYTES + 24576);      // 16 x 512 bf16 probs (M-padded)
  float*  qv    = (float*)(smem + P_LDS_BYTES + 24576 + 16384); // 12 x 16
  float*  qpv   = qv + 192;                                   // 12 x 4 x 3
  float*  hw    = qpv + 144;                                  // 12
  float*  misc  = hw + 12;                                    // R(9), T(3), mask_i(1)

  int i = blockIdx.x;
  int tid = threadIdx.x, lane = tid & 31, wave = tid >> 5;

  // ---- phase 1a: kick off async DMA of p[i] slab into LDS (no VGPR round-trip) ----
  {
    // low 32 bits of the generic LDS pointer == wave-relative LDS byte offset
    unsigned lds_base = (unsigned)(uintptr_t)(void*)p_lds;
    const float* gp = p + (size_t)i * NSEQ * CP;
    for (int c = tid; c < NSEQ * CP / 4; c += 256)            // 16-byte chunks
      async_load_b128_to_lds(lds_base + (unsigned)c * 16u, gp + (size_t)c * 4);
  }

  // ---- phase 1b: per-i query state into LDS (overlaps with the async fill) ----
  for (int idx = tid; idx < NH * CH; idx += 256)
    qv[idx] = proj[(size_t)i * DPROJ + idx];
  for (int idx = tid; idx < NH * PQK * 3; idx += 256)
    qpv[idx] = qp_r[(size_t)i * NH * PQK * 3 + idx];
  if (tid < NH) hw[tid] = logf(1.f + expf(head_w[tid])) * 0.13608276348795434f; // softplus * sqrt(1/54)
  if (tid < 9)  misc[tid] = rot[i * 9 + tid];
  if (tid < 3)  misc[9 + tid] = trans[i * 3 + tid];
  if (tid == 0) misc[12] = mask[i];
  for (int idx = tid; idx < 4 * NSEQ; idx += 256) a_bf[12 * NSEQ + idx] = (__bf16)0.f; // zero M-pad rows
  wait_async_zero();          // each wave drains its own ASYNCcnt...
  __syncthreads();            // ...then workgroup barrier => whole slab visible

  // ---- phase 2: pair bias via WMMA: a_f[h][j] = sqrt(1/3) * p[i,j,:] . w_b[:,h] ----
  for (int jt = wave; jt < NSEQ / 16; jt += 8) {
    v8f acc = {};
    for (int kb = 0; kb < CP; kb += 32) {
      v16bf a = load_a_frag_f32(p_lds, CP, jt * 16, kb, lane);
      v16bf b = load_b_frag(wbT, CP, 0, kb, lane);
      acc = __builtin_amdgcn_wmma_f32_16x16x32_bf16(false, a, false, b, (short)0, acc, false, false);
    }
    int h = lane & 15, roff = (lane >> 4) * 8;
    if (h < NH) {
#pragma unroll
      for (int r = 0; r < 8; ++r)
        a_f[h * NSEQ + jt * 16 + roff + r] = 0.5773502691896258f * acc[r];
    }
  }
  __syncthreads();

  // ---- phase 3: QK dot + point-distance logits + mask (VALU) ----
  {
    float mi = misc[12];
    for (int idx = tid; idx < NSEQ * NH; idx += 256) {
      int j = idx / NH, h = idx % NH;
      const float* krow = proj + (size_t)j * DPROJ + 192 + h * 32;  // k = kv[..., :16]
      float dot = 0.f;
#pragma unroll
      for (int c = 0; c < CH; ++c) dot += qv[h * CH + c] * krow[c];
      const float* kpp = kp_r + ((size_t)j * NH + h) * PQK * 3;
      const float* qpp = qpv + h * PQK * 3;
      float d2 = 0.f;
#pragma unroll
      for (int e = 0; e < PQK * 3; ++e) { float d = qpp[e] - kpp[e]; d2 += d * d; }
      float msk = (mi * mask[j] - 1.f) * 100000.f;
      a_f[h * NSEQ + j] += 0.14433756729740643f * dot - 0.5f * hw[h] * d2 + msk; // sqrt(1/48)
    }
  }
  __syncthreads();

  // ---- phase 4: per-head softmax over j (wave32 shuffle reductions) ----
  for (int h = wave; h < NH; h += 8) {
    float mx = -3.0e38f;
    for (int j = lane; j < NSEQ; j += 32) mx = fmaxf(mx, a_f[h * NSEQ + j]);
    for (int m = 16; m >= 1; m >>= 1) mx = fmaxf(mx, __shfl_xor(mx, m, 32));
    float sum = 0.f;
    for (int j = lane; j < NSEQ; j += 32) {
      float e = __expf(a_f[h * NSEQ + j] - mx);
      a_f[h * NSEQ + j] = e; sum += e;
    }
    for (int m = 16; m >= 1; m >>= 1) sum += __shfl_xor(sum, m, 32);
    float inv = 1.f / sum;
    for (int j = lane; j < NSEQ; j += 32) {
      float pr = a_f[h * NSEQ + j] * inv;
      a_f[h * NSEQ + j] = pr;
      a_bf[h * NSEQ + j] = (__bf16)pr;
    }
  }
  __syncthreads();

  // ---- phase 5a: o_pair = probs @ p[i]  (WMMA, straight from LDS) ----
  {
    int c0 = wave * 16;   // 8 waves cover 128 cols
    v8f acc = {};
    for (int kb = 0; kb < NSEQ; kb += 32) {
      v16bf a = load_a_frag(a_bf, NSEQ, 0, kb, lane);
      v16bf b;
      int c = c0 + (lane & 15);
      int k0 = kb + ((lane >> 4) * 16);
#pragma unroll
      for (int e = 0; e < 16; ++e) b[e] = (__bf16)p_lds[(k0 + e) * CP + c];
      acc = __builtin_amdgcn_wmma_f32_16x16x32_bf16(false, a, false, b, (short)0, acc, false, false);
    }
    int c = c0 + (lane & 15), roff = (lane >> 4) * 8;
#pragma unroll
    for (int r = 0; r < 8; ++r) {
      int h = roff + r;
      if (h < NH) cat_bf[(size_t)i * DCAT + 576 + h * CP + c] = (__bf16)acc[r];
    }
  }

  // ---- phase 5b: o = probs @ v  (VALU, 192 outputs) ----
  for (int idx = tid; idx < NH * CH; idx += 256) {
    int h = idx / CH, c = idx % CH;
    const float* pr = a_f + h * NSEQ;
    float acc = 0.f;
    for (int j = 0; j < NSEQ; ++j)
      acc += pr[j] * proj[(size_t)j * DPROJ + 192 + h * 32 + 16 + c];  // v = kv[..., 16:]
    cat_bf[(size_t)i * DCAT + idx] = (__bf16)acc;
  }
  // ---- phase 5c: op = R_i^T (probs @ vp - T_i), |op| ----
  for (int idx = tid; idx < NH * PV; idx += 256) {
    int h = idx / PV, pp = idx % PV;
    const float* pr = a_f + h * NSEQ;
    float ax = 0.f, ay = 0.f, az = 0.f;
    for (int j = 0; j < NSEQ; ++j) {
      const float* vpj = vp_r + (((size_t)j * NH + h) * PV + pp) * 3;
      float w = pr[j];
      ax += w * vpj[0]; ay += w * vpj[1]; az += w * vpj[2];
    }
    ax -= misc[9]; ay -= misc[10]; az -= misc[11];
    float ox = misc[0] * ax + misc[3] * ay + misc[6] * az;  // R^T
    float oy = misc[1] * ax + misc[4] * ay + misc[7] * az;
    float oz = misc[2] * ax + misc[5] * ay + misc[8] * az;
    size_t base = (size_t)i * DCAT;
    cat_bf[base + 192 + idx * 3 + 0] = (__bf16)ox;
    cat_bf[base + 192 + idx * 3 + 1] = (__bf16)oy;
    cat_bf[base + 192 + idx * 3 + 2] = (__bf16)oz;
    cat_bf[base + 480 + idx] = (__bf16)sqrtf(ox * ox + oy * oy + oz * oz + 1e-8f);
  }
}

// ---------------- layer norm (1 block per row) ----------------
__global__ __launch_bounds__(128) void layer_norm_k(const float* __restrict__ x,
                                                    const float* __restrict__ g,
                                                    const float* __restrict__ b,
                                                    float* __restrict__ out,
                                                    __bf16* __restrict__ out_bf) {
  int row = blockIdx.x, t = threadIdx.x;
  __shared__ float red[128];
  const float* xr = x + (size_t)row * CS;
  float s = 0.f;
  for (int c = t; c < CS; c += 128) s += xr[c];
  red[t] = s; __syncthreads();
  for (int off = 64; off > 0; off >>= 1) { if (t < off) red[t] += red[t + off]; __syncthreads(); }
  float mean = red[0] / CS; __syncthreads();
  float v = 0.f;
  for (int c = t; c < CS; c += 128) { float d = xr[c] - mean; v += d * d; }
  red[t] = v; __syncthreads();
  for (int off = 64; off > 0; off >>= 1) { if (t < off) red[t] += red[t + off]; __syncthreads(); }
  float inv = rsqrtf(red[0] / CS + 1e-5f);
  for (int c = t; c < CS; c += 128) {
    float o = (xr[c] - mean) * inv * g[c] + b[c];
    out[(size_t)row * CS + c] = o;
    if (out_bf) out_bf[(size_t)row * CS + c] = (__bf16)o;
  }
}

// ---------------- backbone update head ----------------
__global__ void final_head_k(const float* __restrict__ s_out,
                             const float* __restrict__ bb_w, const float* __restrict__ bb_b,
                             const float* __restrict__ rot, const float* __restrict__ trans,
                             float* __restrict__ rot_out, float* __restrict__ trans_out) {
  int n = blockIdx.x, t = threadIdx.x;
  __shared__ float upd[6];
  if (t < 6) {
    float acc = bb_b[t];
    const float* sr = s_out + (size_t)n * CS;
    for (int k = 0; k < CS; ++k) acc += sr[k] * bb_w[k * 6 + t];
    upd[t] = acc;
  }
  __syncthreads();
  if (t == 0) {
    float w = 1.f, x = upd[0], y = upd[1], z = upd[2];
    float inv = rsqrtf(w * w + x * x + y * y + z * z);
    w *= inv; x *= inv; y *= inv; z *= inv;
    float U[9] = {1 - 2*(y*y + z*z), 2*(x*y - w*z),     2*(x*z + w*y),
                  2*(x*y + w*z),     1 - 2*(x*x + z*z), 2*(y*z - w*x),
                  2*(x*z - w*y),     2*(y*z + w*x),     1 - 2*(x*x + y*y)};
    const float* R = rot + n * 9;
    float* Ro = rot_out + n * 9;
    for (int ii = 0; ii < 3; ++ii)
      for (int jj = 0; jj < 3; ++jj)
        Ro[ii * 3 + jj] = R[ii * 3 + 0] * U[jj] + R[ii * 3 + 1] * U[3 + jj] + R[ii * 3 + 2] * U[6 + jj];
    float tx = upd[3], ty = upd[4], tz = upd[5];
    const float* T = trans + n * 3;
    float* To = trans_out + n * 3;
    To[0] = R[0] * tx + R[1] * ty + R[2] * tz + T[0];
    To[1] = R[3] * tx + R[4] * ty + R[5] * tz + T[1];
    To[2] = R[6] * tx + R[7] * ty + R[8] * tz + T[2];
  }
}

// ---------------- host launcher ----------------
extern "C" void kernel_launch(void* const* d_in, const int* in_sizes, int n_in,
                              void* d_out, int out_size, void* d_ws, size_t ws_size,
                              hipStream_t stream) {
  const float* s      = (const float*)d_in[0];
  const float* p      = (const float*)d_in[1];
  const float* rot    = (const float*)d_in[2];
  const float* trans  = (const float*)d_in[3];
  const float* mask   = (const float*)d_in[4];
  const float* w_q    = (const float*)d_in[5];
  const float* b_q    = (const float*)d_in[6];
  const float* w_kv   = (const float*)d_in[7];
  const float* b_kv   = (const float*)d_in[8];
  const float* w_qp   = (const float*)d_in[9];
  const float* b_qp   = (const float*)d_in[10];
  const float* w_kvp  = (const float*)d_in[11];
  const float* b_kvp  = (const float*)d_in[12];
  const float* w_b    = (const float*)d_in[13];
  const float* b_b    = (const float*)d_in[14];
  const float* head_w = (const float*)d_in[15];
  const float* w_o    = (const float*)d_in[16];
  const float* b_o    = (const float*)d_in[17];
  const float* ln1_g  = (const float*)d_in[18];
  const float* ln1_b  = (const float*)d_in[19];
  const float* tw1    = (const float*)d_in[20];
  const float* tb1    = (const float*)d_in[21];
  const float* tw2    = (const float*)d_in[22];
  const float* tb2    = (const float*)d_in[23];
  const float* tw3    = (const float*)d_in[24];
  const float* tb3    = (const float*)d_in[25];
  const float* ln2_g  = (const float*)d_in[26];
  const float* ln2_b  = (const float*)d_in[27];
  const float* bb_w   = (const float*)d_in[28];
  const float* bb_b   = (const float*)d_in[29];
  float* out = (float*)d_out;
  float* out_s     = out;                       // 512*384
  float* out_rot   = out + NSEQ * CS;           // 512*9
  float* out_trans = out_rot + NSEQ * 9;        // 512*3

  // workspace carve-out
  char* w = (char*)d_ws;
  auto alloc = [&](size_t bytes) -> void* {
    void* q = (void*)w; w += (bytes + 255) & ~(size_t)255; return q;
  };
  __bf16* s_bf    = (__bf16*)alloc((size_t)NSEQ * CS * 2);
  __bf16* WcatT   = (__bf16*)alloc((size_t)DPROJ * CS * 2);
  float*  bcat    = (float*) alloc(DPROJ * 4);
  __bf16* wbT     = (__bf16*)alloc(16 * CP * 2);
  __bf16* woT     = (__bf16*)alloc((size_t)CS * DCAT * 2);
  __bf16* tw1T    = (__bf16*)alloc((size_t)CS * CS * 2);
  __bf16* tw2T    = (__bf16*)alloc((size_t)CS * CS * 2);
  __bf16* tw3T    = (__bf16*)alloc((size_t)CS * CS * 2);
  float*  proj    = (float*) alloc((size_t)NSEQ * DPROJ * 4);
  float*  qp_r    = (float*) alloc((size_t)NSEQ * NH * PQK * 3 * 4);
  float*  kp_r    = (float*) alloc((size_t)NSEQ * NH * PQK * 3 * 4);
  float*  vp_r    = (float*) alloc((size_t)NSEQ * NH * PV * 3 * 4);
  __bf16* cat_bf  = (__bf16*)alloc((size_t)NSEQ * DCAT * 2);
  float*  s1      = (float*) alloc((size_t)NSEQ * CS * 4);
  float*  sA      = (float*) alloc((size_t)NSEQ * CS * 4);
  __bf16* sA_bf   = (__bf16*)alloc((size_t)NSEQ * CS * 2);
  float*  h1      = (float*) alloc((size_t)NSEQ * CS * 4);
  __bf16* h1_bf   = (__bf16*)alloc((size_t)NSEQ * CS * 2);
  float*  h2      = (float*) alloc((size_t)NSEQ * CS * 4);
  __bf16* h2_bf   = (__bf16*)alloc((size_t)NSEQ * CS * 2);
  float*  s2      = (float*) alloc((size_t)NSEQ * CS * 4);
  (void)ws_size; (void)n_in; (void)in_sizes; (void)out_size;

  auto blocks = [](int count) { return (count + 255) / 256; };

  // 1) weight packing (fp32 -> bf16, B^T layout)
  convert_to_bf16_k<<<blocks(NSEQ * CS), 256, 0, stream>>>(s, s_bf, NSEQ * CS);
  transpose_to_bf16_k<<<blocks(192 * CS), 256, 0, stream>>>(w_q,   WcatT,             CS, 192, 192);
  transpose_to_bf16_k<<<blocks(384 * CS), 256, 0, stream>>>(w_kv,  WcatT + 192 * CS,  CS, 384, 384);
  transpose_to_bf16_k<<<blocks(144 * CS), 256, 0, stream>>>(w_qp,  WcatT + 576 * CS,  CS, 144, 144);
  transpose_to_bf16_k<<<blocks(432 * CS), 256, 0, stream>>>(w_kvp, WcatT + 720 * CS,  CS, 432, 432);
  concat_bias_k<<<blocks(DPROJ), 256, 0, stream>>>(b_q, b_kv, b_qp, b_kvp, bcat);
  transpose_to_bf16_k<<<blocks(16 * CP), 256, 0, stream>>>(w_b, wbT, CP, NH, 16);
  transpose_to_bf16_k<<<blocks(CS * DCAT), 256, 0, stream>>>(w_o, woT, DCAT, CS, CS);
  transpose_to_bf16_k<<<blocks(CS * CS), 256, 0, stream>>>(tw1, tw1T, CS, CS, CS);
  transpose_to_bf16_k<<<blocks(CS * CS), 256, 0, stream>>>(tw2, tw2T, CS, CS, CS);
  transpose_to_bf16_k<<<blocks(CS * CS), 256, 0, stream>>>(tw3, tw3T, CS, CS, CS);

  // 2) fused input projections: proj = s @ [w_q|w_kv|w_qp|w_kvp] + bcat   (512 x 1152, K=384)
  gemm_bf16_k<<<dim3(NSEQ / 16, DPROJ / 128), 256, 0, stream>>>(
      s_bf, WcatT, bcat, nullptr, proj, nullptr, NSEQ, DPROJ, CS, 0);

  // 3) rotate/translate point projections
  rotate_points_k<<<NSEQ, 192, 0, stream>>>(proj, rot, trans, qp_r, kp_r, vp_r);

  // 4) fused IPA attention (async-DMA'd LDS-resident p[i] slab; WMMA bias + o_pair)
  ipa_attn_k<<<NSEQ, 256, ATTN_SMEM, stream>>>(proj, p, wbT, rot, trans, mask, head_w,
                                               qp_r, kp_r, vp_r, cat_bf);

  // 5) output projection + residual: s1 = s + cat @ w_o + b_o   (512 x 384, K=2112)
  gemm_bf16_k<<<dim3(NSEQ / 16, 3), 256, 0, stream>>>(
      cat_bf, woT, b_o, s, s1, nullptr, NSEQ, CS, DCAT, 0);
  layer_norm_k<<<NSEQ, 128, 0, stream>>>(s1, ln1_g, ln1_b, sA, sA_bf);

  // 6) transition MLP
  gemm_bf16_k<<<dim3(NSEQ / 16, 3), 256, 0, stream>>>(
      sA_bf, tw1T, tb1, nullptr, h1, h1_bf, NSEQ, CS, CS, 1);
  gemm_bf16_k<<<dim3(NSEQ / 16, 3), 256, 0, stream>>>(
      h1_bf, tw2T, tb2, nullptr, h2, h2_bf, NSEQ, CS, CS, 1);
  gemm_bf16_k<<<dim3(NSEQ / 16, 3), 256, 0, stream>>>(
      h2_bf, tw3T, tb3, sA, s2, nullptr, NSEQ, CS, CS, 0);
  layer_norm_k<<<NSEQ, 128, 0, stream>>>(s2, ln2_g, ln2_b, out_s, nullptr);

  // 7) backbone update (quat compose + translation)
  final_head_k<<<NSEQ, 32, 0, stream>>>(out_s, bb_w, bb_b, rot, trans, out_rot, out_trans);
}
